// Graph_Transformer_13778255086223
// MI455X (gfx1250) — matrix operand order repaired and checked by reading, not compiled, not run on hardware
//
#include <hip/hip_runtime.h>
#include <math.h>

// ---------------------------------------------------------------------------
// Graph Transformer (TransformerConv x5 + TopKPool x4 + MLP) for gfx1250.
// Dense GEMMs: v_wmma_f32_16x16x32_bf16, wave32, 16x64 output strip per wave
// (4 accumulators, A-fragment reuse), K fully unrolled via templates.
// Weights pre-packed to bf16 in per-lane fragment order -> B loads are two
// contiguous global_load_b128 per fragment instead of 16 strided b32 gathers.
// Scatter softmax / aggregation use global float atomics.
// ---------------------------------------------------------------------------

typedef __attribute__((ext_vector_type(16))) __bf16 bf16x16;
typedef __attribute__((ext_vector_type(8)))  float  f32x8;

#define NGRAPH 128
#define NHEAD  3
#define HEADC  64
#define HC     192   // NHEAD*HEADC
#define EMBD   64
#define EDIM   16

// ------------------------- small helpers -----------------------------------
__device__ __forceinline__ unsigned fkey(float f) {
  unsigned u = __float_as_uint(f);
  return (u & 0x80000000u) ? ~u : (u | 0x80000000u);   // order-preserving key
}
__device__ __forceinline__ float funkey(unsigned k) {
  return __uint_as_float((k & 0x80000000u) ? (k & 0x7FFFFFFFu) : ~k);
}

__global__ void fill_f32(float* p, float v, size_t n) {
  size_t i = (size_t)blockIdx.x * blockDim.x + threadIdx.x;
  if (i < n) p[i] = v;
}
__global__ void fill_u32(unsigned* p, unsigned v, size_t n) {
  size_t i = (size_t)blockIdx.x * blockDim.x + threadIdx.x;
  if (i < n) p[i] = v;
}
__global__ void copy_edges(const int* eidx, int* src, int* dst, int* mask, int E) {
  int e = blockIdx.x * blockDim.x + threadIdx.x;
  if (e >= E) return;
  src[e] = eidx[e];
  dst[e] = eidx[E + e];
  mask[e] = 1;
}

// ------------------------- weight packing -----------------------------------
// Repack W[K,N] (f32, row-major) into bf16 B-fragments:
//   Wpk[((nt*(K/32) + kt)*32 + lane)*16 + e] = W[kt*32 + e + (lane>=16?16:0)][nt*16 + (lane&15)]
// Each (nt,kt) fragment = 32 lanes x 16 bf16 = 1 KB, per-lane chunk 32 B
// contiguous -> GEMM loads it with two global_load_b128.
__global__ void pack_w(const float* __restrict__ W, __bf16* __restrict__ Wpk,
                       int K, int N) {
  int idx = blockIdx.x * blockDim.x + threadIdx.x;
  if (idx >= K * N) return;
  int f    = idx >> 9;         // fragment id (512 elements each)
  int rem  = idx & 511;
  int lane = rem >> 4;
  int e    = rem & 15;
  int kt   = f % (K >> 5);
  int nt   = f / (K >> 5);
  int kb   = e + ((lane >> 4) ? 16 : 0);    // packed B k order (same as A): lanes 0-15 K=e, 16-31 K=e+16
  int l    = lane & 15;
  Wpk[idx] = (__bf16)W[(size_t)(kt * 32 + kb) * N + nt * 16 + l];
}

// ------------------------- WMMA GEMM ----------------------------------------
// C[M,N] = act( A[M,K] @ W[K,N] + bias[N] ), f32 in/out, bf16 WMMA, f32 acc.
// block = (32,4): 4 waves; each wave computes a 16x64 strip (4 x 16x16 tiles),
// reusing its A fragment across the 4 pre-packed B tiles. K is a compile-time
// constant so the K-loop fully unrolls (K/32 steps x 4 WMMA per step).
// grid = (N/64, M/64).
template <int K>
__global__ void wmma_gemm_t(const float* __restrict__ A,
                            const bf16x16* __restrict__ Wf,  // packed fragments
                            const float* __restrict__ bias,
                            float* __restrict__ C,
                            int M, int N, int relu) {
  const int lane = threadIdx.x;          // 0..31 (wave32)
  const int half = lane >> 4;            // 0 / 1
  const int l    = lane & 15;
  const int n0   = blockIdx.x * 64;
  const int m0   = (blockIdx.y * 4 + threadIdx.y) * 16;
  if (m0 >= M) return;                   // wave-uniform exit (EXEC stays full)

  f32x8 acc[4];
#pragma unroll
  for (int t = 0; t < 4; ++t) acc[t] = (f32x8){0.f,0.f,0.f,0.f,0.f,0.f,0.f,0.f};

  const float* arow = A + (size_t)(m0 + l) * K;

#pragma unroll
  for (int k0 = 0; k0 < K; k0 += 32) {
    // A 16x32 bf16 fragment: element e -> K = e + half*16 (matches packed B)
    bf16x16 a;
#pragma unroll
    for (int e = 0; e < 16; ++e) {
      int ka = e + (half ? 16 : 0);
      a[e] = (__bf16)arow[k0 + ka];
    }
#pragma unroll
    for (int t = 0; t < 4; ++t) {
      bf16x16 b = Wf[((size_t)(n0 / 16 + t) * (K >> 5) + (k0 >> 5)) * 32 + lane];
      acc[t] = __builtin_amdgcn_wmma_f32_16x16x32_bf16(
          /*neg_a=*/false, a, /*neg_b=*/false, b,
          /*c_mod=*/(short)0, acc[t], /*reuse_a=*/false, /*reuse_b=*/false);
    }
  }

#pragma unroll
  for (int t = 0; t < 4; ++t) {
    const float bv = bias ? bias[n0 + t * 16 + l] : 0.0f;
#pragma unroll
    for (int r = 0; r < 8; ++r) {
      int m = m0 + r + (half ? 8 : 0);   // D layout: VGPR r -> M=r / M=r+8
      float v = acc[t][r] + bv;
      if (relu) v = fmaxf(v, 0.0f);
      C[(size_t)m * N + (n0 + t * 16 + l)] = v;
    }
  }
}

static inline int cdiv(long long a, long long b) { return (int)((a + b - 1) / b); }

// pack weights into wpk (reused buffer; stream-ordered), then run WMMA GEMM.
static void launch_gemm(const float* A, const float* W, const float* bias,
                        float* C, int M, int K, int N, int relu,
                        __bf16* wpk, hipStream_t st) {
  pack_w<<<cdiv((long long)K * N, 256), 256, 0, st>>>(W, wpk, K, N);
  const bf16x16* Wf = (const bf16x16*)wpk;
  dim3 blk(32, 4);
  dim3 grd(N / 64, M / 64);
  switch (K) {
    case 64:  wmma_gemm_t<64><<<grd, blk, 0, st>>>(A, Wf, bias, C, M, N, relu); break;
    case 128: wmma_gemm_t<128><<<grd, blk, 0, st>>>(A, Wf, bias, C, M, N, relu); break;
    case 192: wmma_gemm_t<192><<<grd, blk, 0, st>>>(A, Wf, bias, C, M, N, relu); break;
    case 256: wmma_gemm_t<256><<<grd, blk, 0, st>>>(A, Wf, bias, C, M, N, relu); break;
    default:  break;  // all K in this model are covered above
  }
}

// ------------------------- edge kernels -------------------------------------
// thread t = edge*3 + head
__global__ void edge_logits(const float* __restrict__ q,
                            const float* __restrict__ kmat,
                            const float* __restrict__ attr,
                            const float* __restrict__ We,
                            const int* __restrict__ src,
                            const int* __restrict__ dst,
                            const int* __restrict__ mask,
                            float* __restrict__ logits,
                            unsigned* __restrict__ maxkey, int E) {
  int t = blockIdx.x * blockDim.x + threadIdx.x;
  if (t >= E * NHEAD) return;
  int e = t / NHEAD, h = t % NHEAD;
  if (!mask[e]) { logits[t] = -3.0e38f; return; }
  int s = src[e], d = dst[e];
  const float* qp = q    + (size_t)d * HC + h * HEADC;
  const float* kp = kmat + (size_t)s * HC + h * HEADC;
  float av[EDIM];
  const float* ap = attr + (size_t)e * EDIM;
#pragma unroll
  for (int j = 0; j < EDIM; ++j) av[j] = ap[j];
  float acc = 0.0f;
  for (int c = 0; c < HEADC; ++c) {
    const float* wc = We + h * HEADC + c;   // column, stride HC
    float ev = 0.0f;
#pragma unroll
    for (int j = 0; j < EDIM; ++j) ev += av[j] * wc[j * HC];
    acc += qp[c] * (kp[c] + ev);
  }
  acc *= 0.125f;                            // 1/sqrt(64)
  logits[t] = acc;
  atomicMax(&maxkey[(size_t)d * NHEAD + h], fkey(acc));
}

__global__ void edge_exp(const float* __restrict__ logits,
                         const int* __restrict__ dst,
                         const int* __restrict__ mask,
                         const unsigned* __restrict__ maxkey,
                         float* __restrict__ ex,
                         float* __restrict__ den, int E) {
  int t = blockIdx.x * blockDim.x + threadIdx.x;
  if (t >= E * NHEAD) return;
  int e = t / NHEAD, h = t % NHEAD;
  if (!mask[e]) { ex[t] = 0.0f; return; }
  int d = dst[e];
  unsigned mk = maxkey[(size_t)d * NHEAD + h];
  float m = (mk == 0u) ? 0.0f : funkey(mk);  // node w/ no in-edge -> m=0
  float v = expf(logits[t] - m);
  ex[t] = v;
  atomicAdd(&den[(size_t)d * NHEAD + h], v);
}

__global__ void edge_agg(const float* __restrict__ ex,
                         const float* __restrict__ den,
                         const float* __restrict__ vmat,
                         const float* __restrict__ attr,
                         const float* __restrict__ We,
                         const int* __restrict__ src,
                         const int* __restrict__ dst,
                         const int* __restrict__ mask,
                         float* __restrict__ agg, int E) {
  int t = blockIdx.x * blockDim.x + threadIdx.x;
  if (t >= E * NHEAD) return;
  int e = t / NHEAD, h = t % NHEAD;
  if (!mask[e]) return;
  float x = ex[t];
  if (x == 0.0f) return;
  int s = src[e], d = dst[e];
  float alpha = x / fmaxf(den[(size_t)d * NHEAD + h], 1e-16f);
  float av[EDIM];
  const float* ap = attr + (size_t)e * EDIM;
#pragma unroll
  for (int j = 0; j < EDIM; ++j) av[j] = ap[j];
  const float* vp = vmat + (size_t)s * HC + h * HEADC;
  float* op = agg + (size_t)d * HC + h * HEADC;
  for (int c = 0; c < HEADC; ++c) {
    const float* wc = We + h * HEADC + c;
    float ev = 0.0f;
#pragma unroll
    for (int j = 0; j < EDIM; ++j) ev += av[j] * wc[j * HC];
    atomicAdd(&op[c], alpha * (vp[c] + ev));
  }
}

// beta-gated skip:  x = b*skip + (1-b)*out,  b = sigmoid([out,skip,out-skip]@Wb)
__global__ void beta_gate(const float* __restrict__ agg,
                          const float* __restrict__ skip,
                          const float* __restrict__ Wb,
                          float* __restrict__ xconv, int NV) {
  int i = blockIdx.x * blockDim.x + threadIdx.x;
  if (i >= NV) return;
  const float* o = agg  + (size_t)i * HC;
  const float* s = skip + (size_t)i * HC;
  float z = 0.0f;
  for (int j = 0; j < HC; ++j) {
    float ov = o[j], sv = s[j];
    z += ov * Wb[j] + sv * Wb[HC + j] + (ov - sv) * Wb[2 * HC + j];
  }
  float b = 1.0f / (1.0f + expf(-z));
  float* xo = xconv + (size_t)i * HC;
  for (int j = 0; j < HC; ++j) xo[j] = b * s[j] + (1.0f - b) * o[j];
}

// ------------------------- batch norm ---------------------------------------
__global__ void bn_stats(const float* __restrict__ x, float* __restrict__ sums,
                         int NV) {
  int t = blockIdx.x * blockDim.x + threadIdx.x;  // 64 blocks * 256
  int c = t & 63, slice = t >> 6;                  // 256 slices per column
  float s = 0.f, s2 = 0.f;
  for (int i = slice; i < NV; i += 256) {
    float v = x[(size_t)i * EMBD + c];
    s += v; s2 += v * v;
  }
  atomicAdd(&sums[c], s);
  atomicAdd(&sums[EMBD + c], s2);
}
__global__ void bn_apply(float* __restrict__ x, const float* __restrict__ sums,
                         const float* __restrict__ gamma,
                         const float* __restrict__ beta, int NV) {
  size_t t = (size_t)blockIdx.x * blockDim.x + threadIdx.x;
  if (t >= (size_t)NV * EMBD) return;
  int c = (int)(t & 63);
  float mu  = sums[c] / NV;
  float var = sums[EMBD + c] / NV - mu * mu;
  x[t] = gamma[c] * (x[t] - mu) * rsqrtf(var + 1e-5f) + beta[c];
}

// ------------------------- TopK pooling -------------------------------------
// one block (256 thr) per graph; keep top n/2 by score = x.p/||p||, scale by
// tanh(score); new_id maps old local node -> new global id (or -1).
__global__ void topk_pool(const float* __restrict__ x,
                          const float* __restrict__ pvec,
                          float* __restrict__ xout,
                          int* __restrict__ new_id, int n) {
  __shared__ float sc[256];
  __shared__ int used[256];
  __shared__ float rv[256];
  __shared__ int ri[256];
  __shared__ float pn;
  int g = blockIdx.x, tid = threadIdx.x;
  if (tid == 0) {
    float s = 0.f;
    for (int j = 0; j < EMBD; ++j) s += pvec[j] * pvec[j];
    pn = sqrtf(s);
  }
  __syncthreads();
  if (tid < n) {
    const float* row = x + (size_t)(g * n + tid) * EMBD;
    float s = 0.f;
    for (int j = 0; j < EMBD; ++j) s += row[j] * pvec[j];
    sc[tid] = s / pn;
    used[tid] = 0;
  }
  for (int i = tid; i < n; i += blockDim.x) new_id[g * n + i] = -1;
  __syncthreads();
  int k = n >> 1;
  for (int r = 0; r < k; ++r) {
    rv[tid] = (tid < n && !used[tid]) ? sc[tid] : -3.4e38f;
    ri[tid] = tid;
    __syncthreads();
    for (int s = 128; s > 0; s >>= 1) {
      if (tid < s && rv[tid + s] > rv[tid]) { rv[tid] = rv[tid + s]; ri[tid] = ri[tid + s]; }
      __syncthreads();
    }
    int idx = ri[0];
    float t = tanhf(sc[idx]);
    const float* row = x + (size_t)(g * n + idx) * EMBD;
    float* orow = xout + (size_t)(g * k + r) * EMBD;
    if (tid < EMBD) orow[tid] = row[tid] * t;
    if (tid == 0) { used[idx] = 1; new_id[g * n + idx] = g * k + r; }
    __syncthreads();
  }
}

__global__ void remap_edges(int* src, int* dst, int* mask,
                            const int* __restrict__ new_id, int E) {
  int e = blockIdx.x * blockDim.x + threadIdx.x;
  if (e >= E) return;
  int m = mask[e];
  int vs = m ? new_id[src[e]] : -1;
  int vd = m ? new_id[dst[e]] : -1;
  int ok = (vs >= 0) && (vd >= 0);
  src[e] = ok ? vs : 0;
  dst[e] = ok ? vd : 0;
  mask[e] = ok;
}

// readout: g[b, 0:64] += max_i x[b,i,:], g[b,64:128] += mean_i x[b,i,:]
__global__ void readout_accum(const float* __restrict__ x,
                              float* __restrict__ gbuf, int n) {
  int t = blockIdx.x * blockDim.x + threadIdx.x;
  if (t >= NGRAPH * EMBD) return;
  int g = t / EMBD, c = t % EMBD;
  float mx = -3.4e38f, sm = 0.f;
  for (int i = 0; i < n; ++i) {
    float v = x[(size_t)(g * n + i) * EMBD + c];
    mx = fmaxf(mx, v); sm += v;
  }
  gbuf[g * 2 * EMBD + c]        += mx;
  gbuf[g * 2 * EMBD + EMBD + c] += sm / n;
}

// final linear, N=2 (not WMMA-tileable)
__global__ void lin3_kernel(const float* __restrict__ h,
                            const float* __restrict__ W,
                            const float* __restrict__ b,
                            float* __restrict__ out) {
  int t = blockIdx.x * blockDim.x + threadIdx.x;
  if (t >= NGRAPH * 2) return;
  int g = t >> 1, c = t & 1;
  float s = b[c];
  for (int j = 0; j < 128; ++j) s += h[g * 128 + j] * W[j * 2 + c];
  out[t] = s;
}

// ---------------------------------------------------------------------------
struct ConvP {
  const float *Wq, *bq, *Wk, *bk, *Wv, *bv, *We, *Wskip, *bskip, *Wbeta;
};
struct Bufs {
  float *q, *k, *v, *skip, *agg, *logits, *ex, *den;
  unsigned* maxkey;
  __bf16* wpk;
};

static void run_conv(const float* x, int NV, int din, const ConvP& p,
                     const float* attr, const int* src, const int* dst,
                     const int* mask, int E, const Bufs& w, hipStream_t st) {
  launch_gemm(x, p.Wq, p.bq, w.q, NV, din, HC, 0, w.wpk, st);
  launch_gemm(x, p.Wk, p.bk, w.k, NV, din, HC, 0, w.wpk, st);
  launch_gemm(x, p.Wv, p.bv, w.v, NV, din, HC, 0, w.wpk, st);
  launch_gemm(x, p.Wskip, p.bskip, w.skip, NV, din, HC, 0, w.wpk, st);

  size_t nvh = (size_t)NV * NHEAD;
  fill_u32<<<cdiv(nvh, 256), 256, 0, st>>>(w.maxkey, 0u, nvh);
  fill_f32<<<cdiv(nvh, 256), 256, 0, st>>>(w.den, 0.f, nvh);
  fill_f32<<<cdiv((size_t)NV * HC, 256), 256, 0, st>>>(w.agg, 0.f, (size_t)NV * HC);

  int EH = E * NHEAD;
  edge_logits<<<cdiv(EH, 256), 256, 0, st>>>(w.q, w.k, attr, p.We, src, dst,
                                             mask, w.logits, w.maxkey, E);
  edge_exp<<<cdiv(EH, 256), 256, 0, st>>>(w.logits, dst, mask, w.maxkey,
                                          w.ex, w.den, E);
  edge_agg<<<cdiv(EH, 256), 256, 0, st>>>(w.ex, w.den, w.v, attr, p.We,
                                          src, dst, mask, w.agg, E);
  // q is dead after edge_logits -> reuse as conv output [NV, HC]
  beta_gate<<<cdiv(NV, 256), 256, 0, st>>>(w.agg, w.skip, p.Wbeta, w.q, NV);
}

extern "C" void kernel_launch(void* const* d_in, const int* in_sizes, int n_in,
                              void* d_out, int out_size, void* d_ws, size_t ws_size,
                              hipStream_t stream) {
  const float* x_in = (const float*)d_in[0];
  const float* attr = (const float*)d_in[1];
  const int* eidx   = (const int*)d_in[2];
  const int E   = in_sizes[2] / 2;        // 262144
  const int NV0 = in_sizes[0] / 128;      // 32768

  auto F = [&](int i) { return (const float*)d_in[i]; };
  auto getconv = [&](int b) {
    ConvP p;
    p.Wq = F(b); p.bq = F(b + 1); p.Wk = F(b + 2); p.bk = F(b + 3);
    p.Wv = F(b + 4); p.bv = F(b + 5); p.We = F(b + 6);
    p.Wskip = F(b + 7); p.bskip = F(b + 8); p.Wbeta = F(b + 9);
    return p;
  };
  // flattened param order (insertion order of setup_inputs dict):
  ConvP conv1 = getconv(4);
  const float* Wt1 = F(14); const float* bt1 = F(15);
  const float* g1 = F(16);  const float* be1 = F(17);
  ConvP convs[4];  for (int i = 0; i < 4; ++i) convs[i] = getconv(18 + 10 * i);
  const float *Wt[4], *bt[4], *bg[4], *bb[4], *pool[4];
  for (int i = 0; i < 4; ++i) { Wt[i] = F(58 + 2 * i); bt[i] = F(59 + 2 * i); }
  for (int i = 0; i < 4; ++i) { bg[i] = F(66 + 2 * i); bb[i] = F(67 + 2 * i); }
  for (int i = 0; i < 4; ++i) pool[i] = F(74 + i);
  const float* Wl1 = F(78); const float* bl1 = F(79);
  const float* Wl2 = F(80); const float* bl2 = F(81);
  const float* Wl3 = F(82); const float* bl3 = F(83);

  // ---- workspace arena ----
  char* wp = (char*)d_ws;
  auto arena = [&](size_t bytes) {
    char* p = wp; wp += (bytes + 255) & ~(size_t)255; return p;
  };
  Bufs w;
  w.q      = (float*)arena((size_t)NV0 * HC * 4);
  w.k      = (float*)arena((size_t)NV0 * HC * 4);
  w.v      = (float*)arena((size_t)NV0 * HC * 4);
  w.skip   = (float*)arena((size_t)NV0 * HC * 4);
  w.agg    = (float*)arena((size_t)NV0 * HC * 4);
  w.logits = (float*)arena((size_t)E * NHEAD * 4);
  w.ex     = (float*)arena((size_t)E * NHEAD * 4);
  w.den    = (float*)arena((size_t)NV0 * NHEAD * 4);
  w.maxkey = (unsigned*)arena((size_t)NV0 * NHEAD * 4);
  w.wpk    = (__bf16*)arena((size_t)256 * 256 * 2);   // max K*N packed bf16
  float* xA     = (float*)arena((size_t)NV0 * EMBD * 4);
  float* xB     = (float*)arena((size_t)NV0 * EMBD * 4);
  int* srcb     = (int*)arena((size_t)E * 4);
  int* dstb     = (int*)arena((size_t)E * 4);
  int* maskb    = (int*)arena((size_t)E * 4);
  int* new_id   = (int*)arena((size_t)NV0 * 4);
  float* gbuf   = (float*)arena((size_t)NGRAPH * 2 * EMBD * 4);
  float* bnsum  = (float*)arena((size_t)2 * EMBD * 4);
  float* h1     = (float*)arena((size_t)NGRAPH * 256 * 4);
  float* h2     = (float*)arena((size_t)NGRAPH * 128 * 4);

  copy_edges<<<cdiv(E, 256), 256, 0, stream>>>(eidx, srcb, dstb, maskb, E);
  fill_f32<<<cdiv(NGRAPH * 2 * EMBD, 256), 256, 0, stream>>>(gbuf, 0.f,
                                                             NGRAPH * 2 * EMBD);

  // ---- conv1 (din=128) + transf1 + bn1 ----
  run_conv(x_in, NV0, 128, conv1, attr, srcb, dstb, maskb, E, w, stream);
  launch_gemm(w.q, Wt1, bt1, xA, NV0, HC, EMBD, 1, w.wpk, stream);
  fill_f32<<<1, 256, 0, stream>>>(bnsum, 0.f, 2 * EMBD);
  bn_stats<<<64, 256, 0, stream>>>(xA, bnsum, NV0);
  bn_apply<<<cdiv((size_t)NV0 * EMBD, 256), 256, 0, stream>>>(xA, bnsum, g1, be1, NV0);

  // ---- 4x (conv + transf + bn + pool + readout) ----
  int n = 256;
  for (int i = 0; i < 4; ++i) {
    int NV = NGRAPH * n;
    run_conv(xA, NV, EMBD, convs[i], attr, srcb, dstb, maskb, E, w, stream);
    launch_gemm(w.q, Wt[i], bt[i], xB, NV, HC, EMBD, 1, w.wpk, stream);
    fill_f32<<<1, 256, 0, stream>>>(bnsum, 0.f, 2 * EMBD);
    bn_stats<<<64, 256, 0, stream>>>(xB, bnsum, NV);
    bn_apply<<<cdiv((size_t)NV * EMBD, 256), 256, 0, stream>>>(xB, bnsum,
                                                               bg[i], bb[i], NV);
    topk_pool<<<NGRAPH, 256, 0, stream>>>(xB, pool[i], xA, new_id, n);
    remap_edges<<<cdiv(E, 256), 256, 0, stream>>>(srcb, dstb, maskb, new_id, E);
    n >>= 1;
    readout_accum<<<cdiv(NGRAPH * EMBD, 256), 256, 0, stream>>>(xA, gbuf, n);
  }

  // ---- MLP head ----
  launch_gemm(gbuf, Wl1, bl1, h1, NGRAPH, 128, 256, 1, w.wpk, stream);
  launch_gemm(h1, Wl2, bl2, h2, NGRAPH, 256, 128, 1, w.wpk, stream);
  lin3_kernel<<<1, 256, 0, stream>>>(h2, Wl3, bl3, (float*)d_out);
}
